// ClusterClsWithSeed_32555852103945
// MI455X (gfx1250) — compile-verified, matrix operand order checked by reading, not run on hardware
//
#include <hip/hip_runtime.h>
#include <stdint.h>

#define HH 1024
#define WW 2048
#define NPIX (HH*WW)
#define NBLK 64
#define TPB  256
#define NTHR (NBLK*TPB)
#define GRP  32            // 4-pixel groups per thread: 32*4*16384 = 2097152
#define CHUNKS (GRP/8)     // 8 packed words per WMMA A operand
#define LN2F 0.6931471805599453f

typedef __attribute__((ext_vector_type(8))) int v8i;

struct GState {
  unsigned bar_cnt, bar_gen;
  unsigned long long amax1, amax2;
  int n1, n2, inter, newUncl, unclSum;
  int count, it;
  int done, stop, brk, big1, accept, idThis;
  float c0, c1, s0, s1;
  int sizes[256];
  int now[256];
  int removeF[256];
};

__device__ __forceinline__ int  ldv (const int* p){ return *(volatile const int*)p; }
__device__ __forceinline__ float ldvf(const float* p){ return *(volatile const float*)p; }
__device__ __forceinline__ unsigned long long ldv64(const unsigned long long* p){ return *(volatile const unsigned long long*)p; }
__device__ __forceinline__ void stv (int* p, int v){ *(volatile int*)p = v; }
__device__ __forceinline__ void stvf(float* p, float v){ *(volatile float*)p = v; }
__device__ __forceinline__ void stv64(unsigned long long* p, unsigned long long v){ *(volatile unsigned long long*)p = v; }

// Sense-reversing grid barrier (64 blocks; each fits a WGP even at max VGPR use).
__device__ __forceinline__ void gsync(GState* S){
  __threadfence();
  __syncthreads();
  if (threadIdx.x == 0){
    volatile unsigned* vg = (volatile unsigned*)&S->bar_gen;
    unsigned old = *vg;
    if (atomicAdd(&S->bar_cnt, 1u) == (unsigned)(NBLK - 1)){
      *(volatile unsigned*)&S->bar_cnt = 0u;
      __threadfence();
      atomicExch(&S->bar_gen, old + 1u);
    } else {
      while (*vg == old) __builtin_amdgcn_s_sleep(1);
    }
  }
  __syncthreads();
  __threadfence();
}

__device__ __forceinline__ int wsum(int v){
  #pragma unroll
  for (int o = 16; o; o >>= 1) v += __shfl_xor(v, o, 32);
  return v;
}
__device__ __forceinline__ unsigned long long wmax64(unsigned long long v){
  #pragma unroll
  for (int o = 16; o; o >>= 1){
    unsigned long long t = __shfl_xor(v, o, 32);
    if (t > v) v = t;
  }
  return v;
}

// Matrix-engine mask popcount: D = A*ones + C.  Per wave, sum(D) == 16 * sum(A bytes),
// with A bytes in {0,1}. One WMMA reduces 1024 mask elements.
__device__ __forceinline__ v8i wmma_cnt(v8i acc, const unsigned* a8){
  v8i a;
  #pragma unroll
  for (int k = 0; k < 8; ++k) a[k] = (int)a8[k];
  v8i b = {0x01010101,0x01010101,0x01010101,0x01010101,
           0x01010101,0x01010101,0x01010101,0x01010101};
  return __builtin_amdgcn_wmma_i32_16x16x64_iu8(false, a, false, b, acc, false, false);
}
__device__ __forceinline__ int vsum8(v8i v){
  int s = 0;
  #pragma unroll
  for (int k = 0; k < 8; ++k) s += v[k];
  return s;
}

__global__ void cluster_init_kernel(GState* S){
  int t = threadIdx.x;
  if (t < 256){ S->sizes[t] = 0; S->now[t] = 0; S->removeF[t] = 0; }
  if (t == 0){
    S->bar_cnt = 0u; S->bar_gen = 0u;
    S->amax1 = 0ull; S->amax2 = 0ull;
    S->n1 = 0; S->n2 = 0; S->inter = 0; S->newUncl = 0; S->unclSum = 0;
    S->count = 1; S->it = 0;
    S->done = 0; S->stop = 0; S->brk = 0; S->big1 = 0; S->accept = 0; S->idThis = 0;
    S->c0 = 0.f; S->c1 = 0.f; S->s0 = 0.f; S->s1 = 0.f;
  }
}

__global__ void __launch_bounds__(TPB)
cluster_main_kernel(const float* __restrict__ P, float* __restrict__ out, GState* S,
                    float* __restrict__ emb0, float* __restrict__ emb1,
                    float* __restrict__ sval, float* __restrict__ smap,
                    uint8_t* __restrict__ uncl, uint8_t* __restrict__ maskb,
                    uint8_t* __restrict__ prop1, uint8_t* __restrict__ prop2,
                    uint8_t* __restrict__ imap){
  const int g = blockIdx.x * TPB + threadIdx.x;           // 0..NTHR-1
  unsigned* unclW  = (unsigned*)uncl;
  unsigned* maskW  = (unsigned*)maskb;
  unsigned* prop1W = (unsigned*)prop1;
  unsigned* prop2W = (unsigned*)prop2;
  unsigned* imapW  = (unsigned*)imap;

  // ---------------- preprocess ----------------
  {
    int mc = 0;
    #pragma unroll 2
    for (int gi = 0; gi < GRP; ++gi){
      int wi = gi * NTHR + g;
      int p  = wi * 4;
      float4 a0 = *(const float4*)(P + 0*NPIX + p);
      float4 a1 = *(const float4*)(P + 1*NPIX + p);
      float4 a4 = *(const float4*)(P + 4*NPIX + p);
      float4 a5 = *(const float4*)(P + 5*NPIX + p);
      float4 a6 = *(const float4*)(P + 6*NPIX + p);
      int row = p >> 11;
      int col = p & (WW - 1);
      float ym = (float)row * (1.0f/1023.0f);
      float4 e0, e1, sv, sm;
      const float xs = 2.0f/2047.0f;
      e0.x = tanhf(a0.x) + (float)(col+0)*xs;
      e0.y = tanhf(a0.y) + (float)(col+1)*xs;
      e0.z = tanhf(a0.z) + (float)(col+2)*xs;
      e0.w = tanhf(a0.w) + (float)(col+3)*xs;
      e1.x = tanhf(a1.x) + ym;  e1.y = tanhf(a1.y) + ym;
      e1.z = tanhf(a1.z) + ym;  e1.w = tanhf(a1.w) + ym;
      sv.x = 1.f/(1.f+expf(-a4.x)); sv.y = 1.f/(1.f+expf(-a4.y));
      sv.z = 1.f/(1.f+expf(-a4.z)); sv.w = 1.f/(1.f+expf(-a4.w));
      sm.x = 1.f/(1.f+expf(-(a6.x-a5.x))); sm.y = 1.f/(1.f+expf(-(a6.y-a5.y)));
      sm.z = 1.f/(1.f+expf(-(a6.z-a5.z))); sm.w = 1.f/(1.f+expf(-(a6.w-a5.w)));
      *(float4*)(emb0 + p) = e0; *(float4*)(emb1 + p) = e1;
      *(float4*)(sval + p) = sv; *(float4*)(smap + p) = sm;
      unsigned f0 = sm.x > 0.5f, f1 = sm.y > 0.5f, f2 = sm.z > 0.5f, f3 = sm.w > 0.5f;
      unsigned pk = f0 | (f1<<8) | (f2<<16) | (f3<<24);
      maskW[wi] = pk; unclW[wi] = pk; imapW[wi] = 0u;
      mc += (int)(f0 + f1 + f2 + f3);
    }
    int s = wsum(mc);
    if ((threadIdx.x & 31) == 0) atomicAdd(&S->unclSum, s);
  }
  gsync(S);

  // ---------------- sequential clustering loop ----------------
  for (;;){
    if (g == 0){
      int c = (!ldv(&S->done)) && (ldv(&S->unclSum) > 160) &&
              (ldv(&S->count) < 200) && (ldv(&S->it) < 2000);
      stv(&S->brk, !c);
      stv64(&S->amax1, 0ull);
    }
    gsync(S);
    if (ldv(&S->brk)) break;

    // pass A: argmax(seed_map * uncl), first-index tie-break via ~idx
    {
      unsigned long long best = 0ull;
      #pragma unroll 2
      for (int gi = 0; gi < GRP; ++gi){
        int wi = gi * NTHR + g; int p = wi * 4;
        float4 sm4 = *(const float4*)(smap + p);
        unsigned uu = unclW[wi];
        const float* smv = (const float*)&sm4;
        #pragma unroll
        for (int j = 0; j < 4; ++j){
          unsigned ub = (uu >> (8*j)) & 1u;
          float v = ub ? smv[j] : 0.0f;
          unsigned long long pk = (((unsigned long long)__float_as_uint(v)) << 32) |
                                  (unsigned long long)(0xFFFFFFFFu - (unsigned)(p + j));
          if (pk > best) best = pk;
        }
      }
      best = wmax64(best);
      if ((threadIdx.x & 31) == 0) atomicMax(&S->amax1, best);
    }
    gsync(S);
    if (g == 0){
      unsigned long long m = ldv64(&S->amax1);
      float v = __uint_as_float((unsigned)(m >> 32));
      if (v < 0.5f){ stv(&S->stop, 1); stv(&S->done, 1); }
      else {
        stv(&S->stop, 0);
        unsigned seed = 0xFFFFFFFFu - (unsigned)m;
        uncl[seed] = 0;
        stvf(&S->c0, emb0[seed]); stvf(&S->c1, emb1[seed]);
        stvf(&S->s0, expf(10.f * P[2*NPIX + seed]));
        stvf(&S->s1, expf(10.f * P[3*NPIX + seed]));
        stv64(&S->amax2, 0ull); stv(&S->n1, 0);
      }
    }
    gsync(S);
    if (ldv(&S->stop)) continue;

    // pass B: proposal 1 -> prop1 bitmap, n1 (WMMA popcount), seed2 argmax(seed_val|prop1).
    // exp(-t) > 0.5  <=>  t < ln2  (dist only feeds the compare; no transcendental needed)
    {
      float c0 = ldvf(&S->c0), c1 = ldvf(&S->c1), s0 = ldvf(&S->s0), s1 = ldvf(&S->s1);
      v8i acc = {0,0,0,0,0,0,0,0};
      unsigned long long best2 = 0ull;
      for (int ch = 0; ch < CHUNKS; ++ch){
        unsigned a8[8];
        #pragma unroll
        for (int j8 = 0; j8 < 8; ++j8){
          int gi = ch*8 + j8; int wi = gi * NTHR + g; int p = wi * 4;
          float4 e0 = *(const float4*)(emb0 + p);
          float4 e1 = *(const float4*)(emb1 + p);
          float4 sv4 = *(const float4*)(sval + p);
          unsigned mk = maskW[wi];
          const float* e0v = (const float*)&e0; const float* e1v = (const float*)&e1;
          const float* svv = (const float*)&sv4;
          unsigned pk = 0;
          #pragma unroll
          for (int j = 0; j < 4; ++j){
            float d0 = e0v[j] - c0, d1 = e1v[j] - c1;
            float t = d0*d0*s0 + d1*d1*s1;
            unsigned f = (t < LN2F) && ((mk >> (8*j)) & 1u);
            pk |= f << (8*j);
            float cv = f ? svv[j] : 0.0f;                 // branch-free candidate
            unsigned long long cand = (((unsigned long long)__float_as_uint(cv)) << 32) |
                                      (unsigned long long)(0xFFFFFFFFu - (unsigned)(p + j));
            if (cand > best2) best2 = cand;
          }
          prop1W[wi] = pk;
          a8[j8] = pk;
        }
        acc = wmma_cnt(acc, a8);
      }
      int s = wsum(vsum8(acc));
      best2 = wmax64(best2);
      if ((threadIdx.x & 31) == 0){
        atomicAdd(&S->n1, s >> 4);
        atomicMax(&S->amax2, best2);
      }
    }
    gsync(S);
    if (g == 0){
      int n1 = ldv(&S->n1);
      int b1 = n1 > 160;
      stv(&S->big1, b1);
      unsigned long long m2 = ldv64(&S->amax2);
      unsigned seed2 = (unsigned)(m2 >> 32) ? (0xFFFFFFFFu - (unsigned)m2) : 0u;
      if (b1) uncl[seed2] = 0;
      stvf(&S->c0, emb0[seed2]); stvf(&S->c1, emb1[seed2]);
      stvf(&S->s0, expf(10.f * P[2*NPIX + seed2]));
      stvf(&S->s1, expf(10.f * P[3*NPIX + seed2]));
      stv(&S->n2, 0); stv(&S->inter, 0);
    }
    gsync(S);

    // pass C: proposal 2 -> prop2 bitmap, n2 and |uncl∧prop2| via two WMMA popcounts
    {
      float c0 = ldvf(&S->c0), c1 = ldvf(&S->c1), s0 = ldvf(&S->s0), s1 = ldvf(&S->s1);
      v8i accN = {0,0,0,0,0,0,0,0};
      v8i accI = {0,0,0,0,0,0,0,0};
      for (int ch = 0; ch < CHUNKS; ++ch){
        unsigned aN[8], aI[8];
        #pragma unroll
        for (int j8 = 0; j8 < 8; ++j8){
          int gi = ch*8 + j8; int wi = gi * NTHR + g; int p = wi * 4;
          float4 e0 = *(const float4*)(emb0 + p);
          float4 e1 = *(const float4*)(emb1 + p);
          unsigned mk = maskW[wi];
          unsigned uu = unclW[wi];
          const float* e0v = (const float*)&e0; const float* e1v = (const float*)&e1;
          unsigned pk = 0;
          #pragma unroll
          for (int j = 0; j < 4; ++j){
            float d0 = e0v[j] - c0, d1 = e1v[j] - c1;
            float t = d0*d0*s0 + d1*d1*s1;
            unsigned f = (t < LN2F) && ((mk >> (8*j)) & 1u);
            pk |= f << (8*j);
          }
          prop2W[wi] = pk;
          aN[j8] = pk;
          aI[j8] = pk & uu;
        }
        accN = wmma_cnt(accN, aN);
        accI = wmma_cnt(accI, aI);
      }
      int sn = wsum(vsum8(accN));
      int si = wsum(vsum8(accI));
      if ((threadIdx.x & 31) == 0){
        atomicAdd(&S->n2, sn >> 4);
        atomicAdd(&S->inter, si >> 4);
      }
    }
    gsync(S);
    if (g == 0){
      int n2 = ldv(&S->n2), inter = ldv(&S->inter);
      int b2 = n2 > 160;
      float ratio = (float)inter / (float)(n2 > 1 ? n2 : 1);
      int acc_ = ldv(&S->big1) && b2 && (ratio > 0.5f);
      stv(&S->accept, acc_);
      if (acc_){
        int c = ldv(&S->count);
        stv(&S->idThis, c);
        S->sizes[c] = n2;
        stv(&S->count, c + 1);
      }
      stv(&S->newUncl, 0);
    }
    gsync(S);

    // pass D: commit imap/uncl, recount uncl via WMMA popcount
    {
      int acc_ = ldv(&S->accept);
      int b1   = ldv(&S->big1);
      unsigned idb = (unsigned)ldv(&S->idThis);
      v8i accU = {0,0,0,0,0,0,0,0};
      for (int ch = 0; ch < CHUNKS; ++ch){
        unsigned a8[8];
        #pragma unroll
        for (int j8 = 0; j8 < 8; ++j8){
          int gi = ch*8 + j8; int wi = gi * NTHR + g;
          unsigned p1 = prop1W[wi], p2 = prop2W[wi];
          unsigned uu = unclW[wi],  im = imapW[wi];
          unsigned fs = b1 ? p2 : p1;
          unsigned nu = 0, ni = 0;
          #pragma unroll
          for (int j = 0; j < 4; ++j){
            unsigned p2b = (p2 >> (8*j)) & 1u;
            unsigned fsb = (fs >> (8*j)) & 1u;
            unsigned ub  = (uu >> (8*j)) & 1u;
            unsigned ib  = (im >> (8*j)) & 0xFFu;
            unsigned ib2 = (acc_ && p2b) ? idb : ib;
            unsigned ub2 = fsb ? 0u : ub;
            nu |= ub2 << (8*j);
            ni |= ib2 << (8*j);
          }
          unclW[wi] = nu; imapW[wi] = ni;
          a8[j8] = nu;
        }
        accU = wmma_cnt(accU, a8);
      }
      int s = wsum(vsum8(accU));
      if ((threadIdx.x & 31) == 0) atomicAdd(&S->newUncl, s >> 4);
    }
    gsync(S);
    if (g == 0){
      stv(&S->unclSum, ldv(&S->newUncl));
      stv(&S->it, ldv(&S->it) + 1);
    }
  }

  // ---------------- post-filter ----------------
  __shared__ int hist[256];
  hist[threadIdx.x] = 0;
  __syncthreads();
  #pragma unroll 2
  for (int gi = 0; gi < GRP; ++gi){
    int wi = gi * NTHR + g;
    unsigned im = imapW[wi];
    #pragma unroll
    for (int j = 0; j < 4; ++j) atomicAdd(&hist[(im >> (8*j)) & 0xFFu], 1);
  }
  __syncthreads();
  {
    int hv = hist[threadIdx.x];
    if (threadIdx.x < 200 && hv) atomicAdd(&S->now[threadIdx.x], hv);
  }
  gsync(S);
  if (blockIdx.x == 0 && threadIdx.x < 200){
    int t  = threadIdx.x;
    int nw = ldv(&S->now[t]);
    int sz = ldv(&S->sizes[t]);
    int rem = (nw != sz) && ((nw < 480) || ((float)nw < 0.5f * (float)sz));
    if (t == 0) rem = 0;
    stv(&S->removeF[t], rem);
  }
  gsync(S);
  #pragma unroll 2
  for (int gi = 0; gi < GRP; ++gi){
    int wi = gi * NTHR + g; int p = wi * 4;
    unsigned im = imapW[wi];
    float4 o;
    float* ov = (float*)&o;
    #pragma unroll
    for (int j = 0; j < 4; ++j){
      unsigned id = (im >> (8*j)) & 0xFFu;
      ov[j] = ldv(&S->removeF[id]) ? 0.f : (float)id;
    }
    *(float4*)(out + p) = o;
  }
}

extern "C" void kernel_launch(void* const* d_in, const int* in_sizes, int n_in,
                              void* d_out, int out_size, void* d_ws, size_t ws_size,
                              hipStream_t stream){
  const float* P = (const float*)d_in[0];
  float* out = (float*)d_out;
  uint8_t* w = (uint8_t*)d_ws;
  size_t off = 4096;                       // GState header (sizeof ~3.2KB)
  GState* S = (GState*)w;
  float* emb0 = (float*)(w + off); off += (size_t)4 * NPIX;
  float* emb1 = (float*)(w + off); off += (size_t)4 * NPIX;
  float* sv   = (float*)(w + off); off += (size_t)4 * NPIX;
  float* sm   = (float*)(w + off); off += (size_t)4 * NPIX;
  uint8_t* uncl  = w + off; off += NPIX;
  uint8_t* maskb = w + off; off += NPIX;
  uint8_t* pr1   = w + off; off += NPIX;
  uint8_t* pr2   = w + off; off += NPIX;
  uint8_t* imap  = w + off; off += NPIX;
  (void)in_sizes; (void)n_in; (void)out_size; (void)ws_size;

  hipLaunchKernelGGL(cluster_init_kernel, dim3(1), dim3(256), 0, stream, S);
  hipLaunchKernelGGL(cluster_main_kernel, dim3(NBLK), dim3(TPB), 0, stream,
                     P, out, S, emb0, emb1, sv, sm, uncl, maskb, pr1, pr2, imap);
}
